// PointnetSAModuleVotes_20014547599434
// MI455X (gfx1250) — matrix operand order repaired and checked by reading, not compiled, not run on hardware
//
#include <hip/hip_runtime.h>
#include <hip/hip_bf16.h>
#include <hip/hip_fp16.h>

typedef __attribute__((ext_vector_type(16))) _Float16 v16h;
typedef __attribute__((ext_vector_type(8)))  _Float16 v8h;
typedef __attribute__((ext_vector_type(8)))  float    v8f;

constexpr int   Bsz = 8, Npts = 16384, Cfeat = 128;
constexpr int   NPOINT = 1024, NSAMPLE = 64;
constexpr float RAD2 = 0.3f * 0.3f;
constexpr int   K0P = 160;            // 131 input channels padded to mult of 32
constexpr int   M0 = 64, M1 = 64, M2 = 128;

// ---------------- output layout (flat, reference return order) ----------------
constexpr size_t OUT_NEWXYZ = 0;                               // B*NPOINT*3 f32
constexpr size_t OUT_FEAT   = (size_t)Bsz * NPOINT * 3;        // B*128*NPOINT f32
constexpr size_t OUT_INDS   = OUT_FEAT + (size_t)Bsz * M2 * NPOINT; // B*NPOINT i32

// ---------------- workspace layout (bytes) ----------------
constexpr size_t WS_IDX = 0;                                            // ball-query idx
constexpr size_t WS_W0  = WS_IDX + (size_t)Bsz * NPOINT * NSAMPLE * 4;  // f16 W0 [64][160]
constexpr size_t WS_W1  = WS_W0 + (size_t)M0 * K0P * 2;                 // f16 W1 [64][64]
constexpr size_t WS_W2  = WS_W1 + (size_t)M1 * M0  * 2;                 // f16 W2 [128][64]

// =====================================================================
// 1) Furthest point sampling: 1 workgroup per batch. Coords staged in
//    dynamic LDS (192KB -- fits CDNA5's 320KB/WGP), running min-dist in
//    registers (64 per thread), LDS tree argmax per iteration.
// =====================================================================
__global__ void fps_kernel(const float* __restrict__ xyz,
                           float* __restrict__ out_newxyz,
                           int*   __restrict__ out_inds) {
  extern __shared__ float smem[];
  float* sx = smem;
  float* sy = smem + Npts;
  float* sz = smem + 2 * Npts;
  float* rv = smem + 3 * Npts;          // 256
  int*   ri = (int*)(rv + 256);         // 256

  const int b = blockIdx.x, t = threadIdx.x;
  const float* xb = xyz + (size_t)b * Npts * 3;
  for (int i = t; i < Npts; i += 256) {
    sx[i] = xb[3 * i + 0];
    sy[i] = xb[3 * i + 1];
    sz[i] = xb[3 * i + 2];
  }
  float mind[Npts / 256];
#pragma unroll
  for (int i = 0; i < Npts / 256; ++i) mind[i] = 1e10f;
  __syncthreads();

  int last = 0;
  if (t == 0) {
    out_inds[(size_t)b * NPOINT] = 0;
    out_newxyz[(size_t)b * NPOINT * 3 + 0] = sx[0];
    out_newxyz[(size_t)b * NPOINT * 3 + 1] = sy[0];
    out_newxyz[(size_t)b * NPOINT * 3 + 2] = sz[0];
  }

  for (int it = 1; it < NPOINT; ++it) {
    const float lx = sx[last], ly = sy[last], lz = sz[last]; // LDS broadcast
    float bestv = -1.0f; int besti = t;
#pragma unroll 4
    for (int i = 0; i < Npts / 256; ++i) {
      const int p = t + i * 256;
      const float dx = sx[p] - lx, dy = sy[p] - ly, dz = sz[p] - lz;
      const float d = dx * dx + dy * dy + dz * dz;
      const float m = fminf(mind[i], d);
      mind[i] = m;
      if (m > bestv) { bestv = m; besti = p; }
    }
    rv[t] = bestv; ri[t] = besti;
    __syncthreads();
    for (int s = 128; s > 0; s >>= 1) {
      if (t < s) {
        const float ov = rv[t + s]; const int oi = ri[t + s];
        if (ov > rv[t] || (ov == rv[t] && oi < ri[t])) { rv[t] = ov; ri[t] = oi; }
      }
      __syncthreads();
    }
    last = ri[0];
    if (t == 0) {
      out_inds[(size_t)b * NPOINT + it] = last;
      out_newxyz[((size_t)b * NPOINT + it) * 3 + 0] = sx[last];
      out_newxyz[((size_t)b * NPOINT + it) * 3 + 1] = sy[last];
      out_newxyz[((size_t)b * NPOINT + it) * 3 + 2] = sz[last];
    }
    __syncthreads();   // protect rv/ri reuse
  }
}

// =====================================================================
// 2) Ball query: one wave32 per center; ballot + prefix-popcount gives
//    order-preserving compaction == "first NSAMPLE indices in ball".
// =====================================================================
__global__ void ballquery_kernel(const float* __restrict__ xyz,
                                 const float* __restrict__ newxyz,
                                 int* __restrict__ idxbuf) {
  const int lane = threadIdx.x & 31;
  const int wv   = threadIdx.x >> 5;
  const int center = blockIdx.x * 8 + wv;          // grid = B*NPOINT/8
  const int b = center >> 10;                      // NPOINT = 1024
  const float* xb = xyz + (size_t)b * Npts * 3;
  const float cx = newxyz[(size_t)center * 3 + 0];
  const float cy = newxyz[(size_t)center * 3 + 1];
  const float cz = newxyz[(size_t)center * 3 + 2];
  int* out = idxbuf + (size_t)center * NSAMPLE;

  int found = 0;
  for (int base = 0; base < Npts && found < NSAMPLE; base += 32) {
    const int p = base + lane;
    const float dx = xb[3 * p + 0] - cx;
    const float dy = xb[3 * p + 1] - cy;
    const float dz = xb[3 * p + 2] - cz;
    const bool in = (dx * dx + dy * dy + dz * dz) < RAD2;
    const unsigned mask = (unsigned)__ballot(in);
    const int rank = __popc(mask & ((1u << lane) - 1u));
    if (in && (found + rank) < NSAMPLE) out[found + rank] = p;
    found += __popc(mask);
  }
  const int start = found < NSAMPLE ? found : NSAMPLE;
  int first = 0;
  if (start > 0) first = out[0];   // same-wave store->load same addr: ordered
  for (int s = start + lane; s < NSAMPLE; s += 32) out[s] = first;
}

// =====================================================================
// 3a) Weight prep: f32 -> f16, K padded to K0P for layer 0.
// =====================================================================
__global__ void convert_weights(const float* __restrict__ W0,
                                const float* __restrict__ W1,
                                const float* __restrict__ W2,
                                _Float16* __restrict__ W0h,
                                _Float16* __restrict__ W1h,
                                _Float16* __restrict__ W2h) {
  const int tid = blockIdx.x * blockDim.x + threadIdx.x;
  const int stride = gridDim.x * blockDim.x;
  for (int i = tid; i < M0 * K0P; i += stride) {
    const int m = i / K0P, k = i % K0P;
    W0h[i] = (_Float16)(k < 131 ? W0[m * 131 + k] : 0.0f);
  }
  for (int i = tid; i < M1 * M0; i += stride) W1h[i] = (_Float16)W1[i];
  for (int i = tid; i < M2 * M1; i += stride) W2h[i] = (_Float16)W2[i];
}

// =====================================================================
// 3b) Fused gather + 3-layer pointwise MLP (WMMA f16->f32) + maxpool.
//     One workgroup (8 wave32) per (batch, center): 64 samples.
//     A frag: weights [16M x 32K] row-major (two 16B global loads/lane)
//     B frag: activations stored pixel-major [s][k] in LDS
//             (one contiguous 32B ds load per lane)
// =====================================================================
template <int K0, int MTILES, int OUTSTRIDE, bool HALF_OUT>
__device__ __forceinline__ void wmma_layer(const _Float16* __restrict__ Wg,
                                           const _Float16* __restrict__ Xs,
                                           void* __restrict__ Ys,
                                           const float* __restrict__ bias,
                                           int lane, int wv) {
  const int col = lane & 15;       // D/B column, also A row
  const int hi  = lane >> 4;       // lane half-group
  const int kbA = hi * 16;
  const int kbB = hi * 16;
  constexpr int NT = MTILES * 4;   // 4 N-tiles of 16 over 64 samples
  for (int t2 = wv; t2 < NT; t2 += 8) {
    const int mt = t2 >> 2, nt = t2 & 3;
    v8f c = {};
#pragma unroll
    for (int kt = 0; kt < K0; kt += 32) {
      const _Float16* wrow = Wg + (size_t)(mt * 16 + col) * K0 + kt + kbA;
      const v8h alo = *(const v8h*)(wrow);
      const v8h ahi = *(const v8h*)(wrow + 8);
      const v16h a = __builtin_shufflevector(alo, ahi, 0, 1, 2, 3, 4, 5, 6, 7,
                                             8, 9, 10, 11, 12, 13, 14, 15);
      const v16h bf = *(const v16h*)(Xs + (size_t)(nt * 16 + col) * K0 + kt + kbB);
      c = __builtin_amdgcn_wmma_f32_16x16x32_f16(false, a, false, bf,
                                                 (short)0, c, false, false);
    }
    const int rbase = mt * 16 + hi * 8;  // D rows = v + 8*hi
    if constexpr (HALF_OUT) {
      v8h o;
#pragma unroll
      for (int v = 0; v < 8; ++v)
        o[v] = (_Float16)fmaxf(c[v] + bias[rbase + v], 0.0f);
      *(v8h*)((_Float16*)Ys + (size_t)(nt * 16 + col) * OUTSTRIDE + rbase) = o;
    } else {
      v8f o;
#pragma unroll
      for (int v = 0; v < 8; ++v)
        o[v] = fmaxf(c[v] + bias[rbase + v], 0.0f);
      *(v8f*)((float*)Ys + (size_t)(nt * 16 + col) * OUTSTRIDE + rbase) = o;
    }
  }
}

__global__ void __launch_bounds__(256)
group_mlp_kernel(const float* __restrict__ xyz,
                 const float* __restrict__ features,
                 const float* __restrict__ newxyz,
                 const int*   __restrict__ idxbuf,
                 const _Float16* __restrict__ W0h,
                 const _Float16* __restrict__ W1h,
                 const _Float16* __restrict__ W2h,
                 const float* __restrict__ b0,
                 const float* __restrict__ b1,
                 const float* __restrict__ b2,
                 float* __restrict__ out_feat) {
  // LDS: a2[64][64] | x0[64][160] | a1[64][64]; f32 pool buffer overlays x0+a1
  __shared__ __align__(32) char smem[40960];
  _Float16* a2   = (_Float16*)(smem);             //     0 ..  8191
  _Float16* x0   = (_Float16*)(smem + 8192);      //  8192 .. 28671
  _Float16* a1   = (_Float16*)(smem + 28672);     // 28672 .. 36863
  float*    outF = (float*)(smem + 8192);         //  8192 .. 40959 (after a1 dead)

  const int t  = threadIdx.x;
  const int bc = blockIdx.x;            // b*NPOINT + c
  const int b  = bc >> 10;
  const int c  = bc & 1023;

  // ---- gather: x0[s][k] = concat(centered xyz, features)[k] at idx[s] ----
  {
    const int s = t & 63;
    const int j = idxbuf[(size_t)bc * NSAMPLE + s];
    const float nx = newxyz[(size_t)bc * 3 + 0];
    const float ny = newxyz[(size_t)bc * 3 + 1];
    const float nz = newxyz[(size_t)bc * 3 + 2];
    const float* pj = xyz + ((size_t)b * Npts + j) * 3;
    for (int k = (t >> 6); k < K0P; k += 4) {
      float v;
      if (k < 3)        v = pj[k] - (k == 0 ? nx : (k == 1 ? ny : nz));
      else if (k < 131) v = features[((size_t)b * Cfeat + (k - 3)) * Npts + j];
      else              v = 0.0f;
      x0[(size_t)s * K0P + k] = (_Float16)v;
    }
  }
  __syncthreads();

  const int lane = t & 31, wv = t >> 5;
  wmma_layer<K0P, M0 / 16, M0, true >(W0h, x0, a1, b0, lane, wv);   // 131->64
  __syncthreads();
  wmma_layer<M0,  M1 / 16, M1, true >(W1h, a1, a2, b1, lane, wv);   // 64->64
  __syncthreads();
  wmma_layer<M1,  M2 / 16, M2, false>(W2h, a2, outF, b2, lane, wv); // 64->128
  __syncthreads();

  // ---- maxpool over 64 samples ----
  if (t < M2) {
    float m = outF[t];
#pragma unroll 8
    for (int s = 1; s < NSAMPLE; ++s) m = fmaxf(m, outF[(size_t)s * M2 + t]);
    out_feat[((size_t)b * M2 + t) * NPOINT + c] = m;
  }
}

// =====================================================================
extern "C" void kernel_launch(void* const* d_in, const int* in_sizes, int n_in,
                              void* d_out, int out_size, void* d_ws, size_t ws_size,
                              hipStream_t stream) {
  const float* xyz      = (const float*)d_in[0];
  const float* features = (const float*)d_in[1];
  const float* W0 = (const float*)d_in[2];
  const float* b0 = (const float*)d_in[3];
  const float* W1 = (const float*)d_in[4];
  const float* b1 = (const float*)d_in[5];
  const float* W2 = (const float*)d_in[6];
  const float* b2 = (const float*)d_in[7];

  float* out        = (float*)d_out;
  float* out_newxyz = out + OUT_NEWXYZ;
  float* out_feat   = out + OUT_FEAT;
  int*   out_inds   = (int*)(out + OUT_INDS);

  char* ws = (char*)d_ws;
  int*      idxbuf = (int*)(ws + WS_IDX);
  _Float16* W0h    = (_Float16*)(ws + WS_W0);
  _Float16* W1h    = (_Float16*)(ws + WS_W1);
  _Float16* W2h    = (_Float16*)(ws + WS_W2);

  // 1) FPS (192KB coords + 2KB reduction in dynamic LDS per workgroup)
  const size_t fps_lds = (size_t)(3 * Npts + 512) * sizeof(float);
  fps_kernel<<<Bsz, 256, fps_lds, stream>>>(xyz, out_newxyz, out_inds);

  // 2) Ball query: 8 waves/block, one wave per center
  ballquery_kernel<<<(Bsz * NPOINT) / 8, 256, 0, stream>>>(xyz, out_newxyz, idxbuf);

  // 3a) Weight conversion to f16 (tiny)
  convert_weights<<<32, 256, 0, stream>>>(W0, W1, W2, W0h, W1h, W2h);

  // 3b) Fused gather + MLP (WMMA) + maxpool: one block per (batch, center)
  group_mlp_kernel<<<Bsz * NPOINT, 256, 0, stream>>>(
      xyz, features, out_newxyz, idxbuf, W0h, W1h, W2h, b0, b1, b2, out_feat);
}
